// MultiHead_72275709657481
// MI455X (gfx1250) — matrix operand order repaired and checked by reading, not compiled, not run on hardware
//
#include <hip/hip_runtime.h>
#include <hip/hip_bf16.h>

#define B_ 4
#define T_ 2048
#define C_ 1024
#define H_ 16
#define D_ 64

typedef __attribute__((ext_vector_type(16))) __bf16 bf16x16;
typedef __attribute__((ext_vector_type(8)))  float  f32x8;

union FragU { uint4 u[2]; bf16x16 v; };

__device__ __forceinline__ f32x8 wmma_bf16(bf16x16 a, bf16x16 b, f32x8 c) {
    // D = A(16x32 bf16) * B(32x16 bf16) + C(16x16 f32)
    return __builtin_amdgcn_wmma_f32_16x16x32_bf16(false, a, false, b, (short)0, c,
                                                   false, false);
}

// A-matrix 16x32 bf16 fragment (ISA 7.12.2): lane = row (0..15); lanes 16-31 same
// rows with kbase=8. Two contiguous 8-elem runs: [kb, kb+8) and [16+kb, 16+kb+8).
__device__ __forceinline__ bf16x16 load_frag_a(const __bf16* p, int stride) {
    const int lane = threadIdx.x & 31;
    const int line = lane & 15;
    const int kb   = (lane >> 4) << 3;   // 0 or 8
    const __bf16* r0 = p + line * stride + kb;
    FragU f;
    f.u[0] = *(const uint4*)(r0);
    f.u[1] = *(const uint4*)(r0 + 16);
    return f.v;
}

// B-matrix 32x16 bf16 fragment: lane = column; lanes 16-31 use kbase=8.
// Two contiguous 8-elem runs along K: [kb, kb+8) and [16+kb, 16+kb+8).
// Element B[k][n] at p[n*stride + k] (column-contiguous storage).
__device__ __forceinline__ bf16x16 load_frag_b(const __bf16* p, int stride) {
    const int lane = threadIdx.x & 31;
    const int line = lane & 15;
    const int kb   = (lane >> 4) << 3;   // 0 or 8
    const __bf16* r0 = p + line * stride + kb;
    FragU f;
    f.u[0] = *(const uint4*)(r0);
    f.u[1] = *(const uint4*)(r0 + 16);
    return f.v;
}

// Fragment set for one 32-wide K step of a 32x64 GEMM tile.
struct FragSet {
    bf16x16 a0, a1, b0, b1, b2, b3;
};

__device__ __forceinline__ FragSet load_set(const __bf16* abase, const __bf16* wt, int kk) {
    FragSet f;
    f.a0 = load_frag_a(abase + kk, C_);
    f.a1 = load_frag_a(abase + (size_t)16 * C_ + kk, C_);
    f.b0 = load_frag_b(wt + kk, C_);
    f.b1 = load_frag_b(wt + (size_t)16 * C_ + kk, C_);
    f.b2 = load_frag_b(wt + (size_t)32 * C_ + kk, C_);
    f.b3 = load_frag_b(wt + (size_t)48 * C_ + kk, C_);
    return f;
}

__device__ __forceinline__ void wmma_set(const FragSet& f, f32x8 acc[8]) {
    acc[0] = wmma_bf16(f.a0, f.b0, acc[0]);
    acc[1] = wmma_bf16(f.a0, f.b1, acc[1]);
    acc[2] = wmma_bf16(f.a0, f.b2, acc[2]);
    acc[3] = wmma_bf16(f.a0, f.b3, acc[3]);
    acc[4] = wmma_bf16(f.a1, f.b0, acc[4]);
    acc[5] = wmma_bf16(f.a1, f.b1, acc[5]);
    acc[6] = wmma_bf16(f.a1, f.b2, acc[6]);
    acc[7] = wmma_bf16(f.a1, f.b3, acc[7]);
}

// Software-pipelined 32x64 GEMM accumulation over K = C_ (ping-pong fragment sets).
// NOTE: issues one dead load-set one step past K; stays inside the workspace.
__device__ __forceinline__ void gemm_32x64(const __bf16* abase, const __bf16* wt,
                                           f32x8 acc[8]) {
    FragSet f0 = load_set(abase, wt, 0);
    for (int i = 0; i < 16; ++i) {
        const int kk = i * 64;
        FragSet f1 = load_set(abase, wt, kk + 32);   // in flight during wmma_set(f0)
        wmma_set(f0, acc);
        f0 = load_set(abase, wt, kk + 64);           // in flight during wmma_set(f1)
        wmma_set(f1, acc);
    }
}

// ---------------- Kernel 1a: f32 -> bf16 convert, 8 elems/thread ----------------
__global__ void k_convert8(const float* __restrict__ in, __bf16* __restrict__ out, int n8) {
    int i = blockIdx.x * blockDim.x + threadIdx.x;
    if (i >= n8) return;
    const float4* p = (const float4*)in + (size_t)i * 2;
    float4 f0 = p[0], f1 = p[1];
    union { __bf16 h[8]; uint4 u; } o;
    o.h[0] = (__bf16)f0.x; o.h[1] = (__bf16)f0.y; o.h[2] = (__bf16)f0.z; o.h[3] = (__bf16)f0.w;
    o.h[4] = (__bf16)f1.x; o.h[5] = (__bf16)f1.y; o.h[6] = (__bf16)f1.z; o.h[7] = (__bf16)f1.w;
    *((uint4*)out + i) = o.u;
}

// ---- Kernel 1b: f32 [batch, 2^lr rows, 2^lc cols] -> bf16 [batch, cols, rows] ----
// Power-of-two dims: pure shift/mask indexing (no integer division).
__global__ void k_transpose_convert(const float* __restrict__ in, __bf16* __restrict__ out,
                                    int lr, int lc, int n) {
    int i = blockIdx.x * blockDim.x + threadIdx.x;
    if (i >= n) return;
    int c = i & ((1 << lc) - 1);
    int r = (i >> lc) & ((1 << lr) - 1);
    int bidx = i >> (lc + lr);
    out[(((size_t)(bidx << lc) + c) << lr) + r] = (__bf16)in[i];
}

// ---------------- Kernel 2: QKV projection ----------------
// grid = (64, 48): y = mat*16 + h (mat: 0=q,1=k,2=v). Each wave: 32 rows x 64 cols.
__global__ void __launch_bounds__(128) k_qkv(const __bf16* __restrict__ xb,
                                             const __bf16* __restrict__ wqt,
                                             const __bf16* __restrict__ wkt,
                                             const __bf16* __restrict__ wvt,
                                             __bf16* __restrict__ qo,
                                             __bf16* __restrict__ ko,
                                             __bf16* __restrict__ vto) {
    const int wave = threadIdx.x >> 5;
    const int lane = threadIdx.x & 31;
    const int mt   = blockIdx.x * 4 + wave;      // 0..255 (32-row tiles over B*T)
    const int y    = blockIdx.y;
    const int mat  = y >> 4;
    const int h    = y & 15;
    const __bf16* wt = (mat == 0 ? wqt : (mat == 1 ? wkt : wvt)) + (size_t)h * D_ * C_;
    const size_t m0 = (size_t)mt * 32;
    f32x8 fz = {0.f,0.f,0.f,0.f,0.f,0.f,0.f,0.f};
    f32x8 acc[8] = {fz, fz, fz, fz, fz, fz, fz, fz};
    gemm_32x64(xb + m0 * C_, wt, acc);

    const int g = lane >> 4, nl = lane & 15;
#pragma unroll
    for (int aI = 0; aI < 8; ++aI) {
        const int sub = aI >> 2;                 // 0: rows m0..+15, 1: rows m0+16..+31
#pragma unroll
        for (int r = 0; r < 8; ++r) {
            int row = (int)m0 + sub * 16 + r + g * 8;   // global row in [B*T]
            int b   = row >> 11;                        // T_ = 2048
            int t   = row & (T_ - 1);
            int d   = (aI & 3) * 16 + nl;
            __bf16 val = (__bf16)acc[aI][r];
            size_t bh = (size_t)b * H_ + h;
            if (mat == 0)      qo[(bh * T_ + t) * D_ + d] = val;      // [B,H,T,D]
            else if (mat == 1) ko[(bh * T_ + t) * D_ + d] = val;      // [B,H,T,D]
            else               vto[(bh * D_ + d) * T_ + t] = val;     // [B,H,D,T]
        }
    }
}

// ---------------- Kernel 3: causal flash attention ----------------
// One wave per 16-row Q tile. grid = 2048 blocks of 128.
__global__ void __launch_bounds__(128) k_attn(const __bf16* __restrict__ q,
                                              const __bf16* __restrict__ k,
                                              const __bf16* __restrict__ vT,
                                              __bf16* __restrict__ attn) {
    __shared__ __align__(16) __bf16 plds[4][16][40];   // P staging, padded stride 40
    const int wave = threadIdx.x >> 5;
    const int lane = threadIdx.x & 31;
    const int wid  = blockIdx.x * 4 + wave;            // 0..8191
    const int tt   = wid & (T_ / 16 - 1);
    const int bh   = wid >> 7;
    const int b    = bh >> 4;
    const int h    = bh & 15;
    const int t0   = tt * 16;
    const int g = lane >> 4, nl = lane & 15;
    const __bf16* qb = q  + (size_t)bh * T_ * D_;
    const __bf16* kb = k  + (size_t)bh * T_ * D_;
    const __bf16* vb = vT + (size_t)bh * D_ * T_;
    // Q fragments stay in registers across all key blocks
    bf16x16 qa0 = load_frag_a(qb + (size_t)t0 * D_, D_);
    bf16x16 qa1 = load_frag_a(qb + (size_t)t0 * D_ + 32, D_);
    f32x8 fz = {0.f,0.f,0.f,0.f,0.f,0.f,0.f,0.f};
    f32x8 o[4] = {fz, fz, fz, fz};
    float m8[8], l8[8];
#pragma unroll
    for (int r = 0; r < 8; ++r) { m8[r] = -1e30f; l8[r] = 0.f; }
    const float scale = 0.03125f;                      // C^-0.5 = 1/32
    const int nblk = (t0 >> 5) + 1;                    // causal: s <= t0+15
    for (int blk = 0; blk < nblk; ++blk) {
        const int s0 = blk << 5;
        // K fragments FIRST (loads complete in order): the QK WMMAs only have to
        // wait for K; V loads stay in flight through the softmax VALU section.
        bf16x16 kf0 = load_frag_b(kb + (size_t)s0 * D_, D_);
        bf16x16 kf1 = load_frag_b(kb + (size_t)s0 * D_ + 32, D_);
        bf16x16 kf2 = load_frag_b(kb + (size_t)(s0 + 16) * D_, D_);
        bf16x16 kf3 = load_frag_b(kb + (size_t)(s0 + 16) * D_ + 32, D_);
        bf16x16 vf0 = load_frag_b(vb + (size_t)0  * T_ + s0, T_);
        bf16x16 vf1 = load_frag_b(vb + (size_t)16 * T_ + s0, T_);
        bf16x16 vf2 = load_frag_b(vb + (size_t)32 * T_ + s0, T_);
        bf16x16 vf3 = load_frag_b(vb + (size_t)48 * T_ + s0, T_);
        // prefetch next key block (uniform addresses; global_prefetch_b8)
        __builtin_prefetch(kb + (size_t)(s0 + 32) * D_, 0, 3);
        __builtin_prefetch(vb + (size_t)(s0 + 32), 0, 3);
        // S = Q * K^T
        f32x8 s1 = wmma_bf16(qa0, kf0, fz);
        s1 = wmma_bf16(qa1, kf1, s1);
        f32x8 s2 = wmma_bf16(qa0, kf2, fz);
        s2 = wmma_bf16(qa1, kf3, s2);
        const int c1 = s0 + nl, c2 = s0 + 16 + nl;
        float e1[8], e2[8];
#pragma unroll
        for (int r = 0; r < 8; ++r) {
            const int trow = t0 + r + g * 8;
            float v1 = (c1 <= trow) ? s1[r] * scale : -1e30f;
            float v2 = (c2 <= trow) ? s2[r] * scale : -1e30f;
            // row max across the 16 lanes holding this row's columns
            float mx = fmaxf(v1, v2);
            mx = fmaxf(mx, __shfl_xor(mx, 1, 32));
            mx = fmaxf(mx, __shfl_xor(mx, 2, 32));
            mx = fmaxf(mx, __shfl_xor(mx, 4, 32));
            mx = fmaxf(mx, __shfl_xor(mx, 8, 32));
            const float newm = fmaxf(m8[r], mx);
            const float corr = __expf(m8[r] - newm);
            m8[r] = newm;
            const float a1 = __expf(v1 - newm);
            const float a2 = __expf(v2 - newm);
            float rs = a1 + a2;
            rs += __shfl_xor(rs, 1, 32);
            rs += __shfl_xor(rs, 2, 32);
            rs += __shfl_xor(rs, 4, 32);
            rs += __shfl_xor(rs, 8, 32);
            l8[r] = l8[r] * corr + rs;
            o[0][r] *= corr; o[1][r] *= corr; o[2][r] *= corr; o[3][r] *= corr;
            e1[r] = a1; e2[r] = a2;
        }
        // stage P (16x32) in LDS to redistribute rows->lanes for the A fragment
#pragma unroll
        for (int r = 0; r < 8; ++r) {
            plds[wave][r + g * 8][nl]      = (__bf16)e1[r];
            plds[wave][r + g * 8][16 + nl] = (__bf16)e2[r];
        }
        asm volatile("s_wait_dscnt 0" ::: "memory");
        bf16x16 pa = load_frag_a(&plds[wave][0][0], 40);
        // O += P * V (V fragments already resident)
        o[0] = wmma_bf16(pa, vf0, o[0]);
        o[1] = wmma_bf16(pa, vf1, o[1]);
        o[2] = wmma_bf16(pa, vf2, o[2]);
        o[3] = wmma_bf16(pa, vf3, o[3]);
    }
    // normalize and write concatenated heads: attn[(b*T + t)*C + h*64 + d]  (bf16)
#pragma unroll
    for (int aI = 0; aI < 4; ++aI) {
#pragma unroll
        for (int r = 0; r < 8; ++r) {
            const int trow = t0 + r + g * 8;
            const int d = h * D_ + aI * 16 + nl;
            attn[((size_t)b * T_ + trow) * C_ + d] = (__bf16)(o[aI][r] / l8[r]);
        }
    }
}

// ---------------- Kernel 4: output projection + bias ----------------
// grid = (64, 16). Each wave: 32 rows x 64 cols, K = C = 1024.
__global__ void __launch_bounds__(128) k_proj(const __bf16* __restrict__ a,
                                              const __bf16* __restrict__ wpt,
                                              const float* __restrict__ bias,
                                              float* __restrict__ out) {
    const int wave = threadIdx.x >> 5;
    const int lane = threadIdx.x & 31;
    const int mt = blockIdx.x * 4 + wave;        // 0..255
    const int n0 = blockIdx.y * 64;
    const size_t m0 = (size_t)mt * 32;
    f32x8 fz = {0.f,0.f,0.f,0.f,0.f,0.f,0.f,0.f};
    f32x8 acc[8] = {fz, fz, fz, fz, fz, fz, fz, fz};
    gemm_32x64(a + m0 * C_, wpt + (size_t)n0 * C_, acc);

    const int g = lane >> 4, nl = lane & 15;
#pragma unroll
    for (int aI = 0; aI < 8; ++aI) {
        const int sub = aI >> 2;
        const float bv = bias[n0 + (aI & 3) * 16 + nl];
#pragma unroll
        for (int r = 0; r < 8; ++r)
            out[(m0 + sub * 16 + r + g * 8) * C_ + n0 + (aI & 3) * 16 + nl] =
                acc[aI][r] + bv;
    }
}

extern "C" void kernel_launch(void* const* d_in, const int* in_sizes, int n_in,
                              void* d_out, int out_size, void* d_ws, size_t ws_size,
                              hipStream_t stream) {
    (void)in_sizes; (void)n_in; (void)out_size; (void)ws_size;
    const float* x      = (const float*)d_in[0];
    const float* wq     = (const float*)d_in[1];
    const float* wk     = (const float*)d_in[2];
    const float* wv     = (const float*)d_in[3];
    const float* w_proj = (const float*)d_in[4];
    const float* b_proj = (const float*)d_in[5];
    float* out = (float*)d_out;

    const size_t NX = (size_t)B_ * T_ * C_;      // 8388608
    const size_t NW = (size_t)H_ * C_ * D_;      // 1048576
    const size_t NP = (size_t)C_ * C_;           // 1048576
    const size_t NQ = (size_t)B_ * H_ * T_ * D_; // 8388608

    __bf16* ws    = (__bf16*)d_ws;
    __bf16* xb    = ws;              // [B*T, C] bf16
    __bf16* wqt   = xb + NX;         // [H, D, C] bf16 (transposed)
    __bf16* wkt   = wqt + NW;
    __bf16* wvt   = wkt + NW;
    __bf16* wpt   = wvt + NW;        // [Cout, Cin] bf16 (transposed w_proj)
    __bf16* qb    = wpt + NP;        // [B,H,T,D]
    __bf16* kb    = qb + NQ;         // [B,H,T,D]
    __bf16* vtb   = kb + NQ;         // [B,H,D,T]
    __bf16* attnb = xb;              // reuse x_bf16 region after QKV

    k_convert8<<<dim3((unsigned)(NX / 8 / 256)), dim3(256), 0, stream>>>(x, xb, (int)(NX / 8));
    // wq/wk/wv: [H, C(2^10) x D(2^6)] -> [H, D, C]
    k_transpose_convert<<<dim3((unsigned)((NW + 255) / 256)), dim3(256), 0, stream>>>(
        wq, wqt, 10, 6, (int)NW);
    k_transpose_convert<<<dim3((unsigned)((NW + 255) / 256)), dim3(256), 0, stream>>>(
        wk, wkt, 10, 6, (int)NW);
    k_transpose_convert<<<dim3((unsigned)((NW + 255) / 256)), dim3(256), 0, stream>>>(
        wv, wvt, 10, 6, (int)NW);
    // w_proj: [C(2^10) x C(2^10)] -> transposed
    k_transpose_convert<<<dim3((unsigned)((NP + 255) / 256)), dim3(256), 0, stream>>>(
        w_proj, wpt, 10, 10, (int)NP);

    k_qkv<<<dim3(64, 48), dim3(128), 0, stream>>>(xb, wqt, wkt, wvt, qb, kb, vtb);
    k_attn<<<dim3(2048), dim3(128), 0, stream>>>(qb, kb, vtb, attnb);
    k_proj<<<dim3(64, 16), dim3(128), 0, stream>>>(attnb, wpt, b_proj, out);
}
